// SwinTransformerBlock_20538533609575
// MI455X (gfx1250) — matrix operand order repaired and checked by reading, not compiled, not run on hardware
//
#include <hip/hip_runtime.h>
#include <hip/hip_bf16.h>
#include <hip/hip_fp16.h>
#include <stdint.h>

// ---------------------------------------------------------------------------
// Swin Transformer block on gfx1250 (MI455X): f16 WMMA GEMMs + WMMA attention
// A-tile staging: TDM tensor_load_to_lds > async-to-LDS > sync fallback
// ---------------------------------------------------------------------------

typedef _Float16 half_t;
typedef __attribute__((ext_vector_type(16))) _Float16 v16h;
typedef __attribute__((ext_vector_type(8)))  _Float16 v8h;
typedef __attribute__((ext_vector_type(8)))  float    v8f;
typedef __attribute__((ext_vector_type(4)))  unsigned int u32x4;
typedef __attribute__((ext_vector_type(8)))  int      i32x8;
typedef __attribute__((ext_vector_type(4)))  int      i32x4;

#define WMMA_F16(a, b, c) \
  __builtin_amdgcn_wmma_f32_16x16x32_f16(false, (a), false, (b), (short)0, (c), false, false)

#define MTOK 200704   // 64 * 3136 tokens

#if __has_builtin(__builtin_amdgcn_tensor_load_to_lds) && \
    __has_builtin(__builtin_amdgcn_s_wait_tensorcnt)
#define STAGE_TDM 1
#else
#define STAGE_TDM 0
#endif
#if __has_builtin(__builtin_amdgcn_global_load_async_to_lds_b128)
#define STAGE_ASYNC 1
#else
#define STAGE_ASYNC 0
#endif

#if __has_builtin(__builtin_amdgcn_s_wait_asynccnt)
#define WAIT_ASYNC0() __builtin_amdgcn_s_wait_asynccnt(0)
#else
#define WAIT_ASYNC0() asm volatile("s_wait_asynccnt 0x0" ::: "memory")
#endif

union H16 { v16h v; v8h h[2]; };

// A-fragment (16x32 f16), row-major source with stride ld (halves).
// lane&15 = row; lanes>=16 take K-offsets +8 / +24 (ISA 7.12.2, 16-bit A 16x32)
__device__ __forceinline__ v16h load_a_frag(const half_t* base, int ld, int lane) {
  const half_t* r = base + (lane & 15) * ld + ((lane & 16) ? 8 : 0);
  H16 u; u.h[0] = *(const v8h*)(r); u.h[1] = *(const v8h*)(r + 16);
  return u.v;
}
// B-fragment (32x16 f16) from a TRANSPOSED store (N x K row-major, stride ld):
// lane&15 = output column; lanes>=16 take K-offsets +8 / +24 (same order as A)
__device__ __forceinline__ v16h load_b_frag(const half_t* base, int ld, int lane) {
  const half_t* r = base + (lane & 15) * ld + ((lane & 16) ? 8 : 0);
  H16 u; u.h[0] = *(const v8h*)(r); u.h[1] = *(const v8h*)(r + 16);
  return u.v;
}
__device__ __forceinline__ v16h load_a_frag_ptr(const half_t* rowptr, int lane) {
  const half_t* r = rowptr + ((lane & 16) ? 8 : 0);
  H16 u; u.h[0] = *(const v8h*)(r); u.h[1] = *(const v8h*)(r + 16);
  return u.v;
}
__device__ __forceinline__ v16h load_b_frag_ptr(const half_t* colptr, int lane) {
  const half_t* r = colptr + ((lane & 16) ? 8 : 0);
  H16 u; u.h[0] = *(const v8h*)(r); u.h[1] = *(const v8h*)(r + 16);
  return u.v;
}

#if STAGE_TDM
// TDM: DMA a 128-row x 32-half tile (row stride K halves) from global into LDS,
// padding 16B after every 64B row -> LDS row stride 40 halves (bank-safe).
// D# per CDNA5 ISA.
__device__ __forceinline__ void tdm_load_tile(const half_t* g, half_t* l, int K) {
  uint64_t ga = (uint64_t)(uintptr_t)g;
  u32x4 g0;
  g0.x = 1u;                                   // count=1, user mode
  g0.y = (uint32_t)(uintptr_t)l;               // lds_addr (wave-relative offset)
  g0.z = (uint32_t)ga;                         // global_addr[31:0]
  g0.w = (uint32_t)((ga >> 32) & 0x01FFFFFFu) | (2u << 30);  // addr[56:32], type=2
  i32x8 g1;
  g1[0] = (1 << 16)      // data_size = 2 bytes
        | (1 << 20)      // pad_enable
        | (3 << 22)      // pad_interval: every 16 DWORDs (64B = one 32-half row)
        | (3 << 25);     // pad_amount: 4 DWORDs (16B = 8 halves) -> stride 40
  g1[1] = (K & 0xFFFF) << 16;                       // tensor_dim0[15:0]
  g1[2] = ((unsigned)K >> 16);                      // tensor_dim0[31:16], dim1 lo=0
  g1[3] = 16 | (32 << 16);                          // tensor_dim1 = 1<<20, tile_dim0=32
  g1[4] = 128;                                      // tile_dim1=128, tile_dim2=0
  g1[5] = K;                                        // tensor_dim0_stride[31:0]
  g1[6] = 0;
  g1[7] = 0;
  i32x4 z4 = {0, 0, 0, 0};
#if __clang_major__ >= 23
  i32x8 z8 = {0, 0, 0, 0, 0, 0, 0, 0};
  __builtin_amdgcn_tensor_load_to_lds(g0, g1, z4, z4, z8, 0);
#else
  __builtin_amdgcn_tensor_load_to_lds(g0, g1, z4, z4, 0);
#endif
}
#endif

#if STAGE_ASYNC
// builtin signature (from clang diagnostic): (i32x4 AS1*, i32x4 AS3*, Ii, Ii)
typedef __attribute__((address_space(1))) i32x4 as1_i32x4;
typedef __attribute__((address_space(3))) i32x4 as3_i32x4;
__device__ __forceinline__ void async_copy16(const half_t* g, half_t* l) {
  __builtin_amdgcn_global_load_async_to_lds_b128(
      (as1_i32x4*)(uintptr_t)g,
      (as3_i32x4*)(uint32_t)(uintptr_t)l,
      0, 0);
}
#endif

// ---------------------------------------------------------------------------
// Weight prep: fp32 (K x N) -> f16 transposed (N x K)
// ---------------------------------------------------------------------------
__global__ __launch_bounds__(256) void convT_kernel(const float* __restrict__ w,
                                                    half_t* __restrict__ wt,
                                                    int K, int N) {
  int idx = blockIdx.x * 256 + threadIdx.x;
  if (idx < N * K) {
    int n = idx / K, k = idx - n * K;
    wt[idx] = (half_t)w[(size_t)k * N + n];
  }
}

// Relative-position-bias table: tab[h][i][j], i,j in 0..48
__global__ __launch_bounds__(256) void biastab_kernel(const float* __restrict__ rpb,
                                                      float* __restrict__ tab) {
  int idx = blockIdx.x * 256 + threadIdx.x;
  if (idx < 12 * 49 * 49) {
    int h = idx / 2401;
    int rem = idx - h * 2401;
    int i = rem / 49, j = rem - i * 49;
    int ri = i / 7, ci = i - ri * 7;
    int rj = j / 7, cj = j - rj * 7;
    int rel = (ri - rj + 6) * 13 + (ci - cj + 6);
    tab[idx] = rpb[rel * 12 + h];
  }
}

// ---------------------------------------------------------------------------
// LayerNorm (+ optional roll/window permute) fp32 -> f16.  One wave per token.
// ---------------------------------------------------------------------------
__global__ __launch_bounds__(256) void ln_kernel(const float* __restrict__ x,
                                                 const float* __restrict__ gam,
                                                 const float* __restrict__ bet,
                                                 half_t* __restrict__ out, int mode) {
  int wave = threadIdx.x >> 5, lane = threadIdx.x & 31;
  size_t t = (size_t)blockIdx.x * 8 + wave;   // destination row
  size_t srcrow;
  if (mode == 0) {
    int wt = (int)t;
    int win = wt / 49, ii = wt - win * 49;
    int b = win >> 6, w = win & 63;
    int r7 = ii / 7, c7 = ii - r7 * 7;
    int gh = (w >> 3) * 7 + r7, gw = (w & 7) * 7 + c7;
    int oh = gh + 3; if (oh >= 56) oh -= 56;   // roll(-3): shifted[gh] = x[(gh+3)%56]
    int ow = gw + 3; if (ow >= 56) ow -= 56;
    srcrow = (size_t)b * 3136 + oh * 56 + ow;
  } else {
    srcrow = t;
  }
  const float* src = x + srcrow * 384;
  float v[12], s = 0.f;
#pragma unroll
  for (int j = 0; j < 12; j++) { v[j] = src[lane + 32 * j]; s += v[j]; }
#pragma unroll
  for (int d = 16; d >= 1; d >>= 1) s += __shfl_xor(s, d, 32);
  float mean = s * (1.0f / 384.0f);
  float q = 0.f;
#pragma unroll
  for (int j = 0; j < 12; j++) { float dv = v[j] - mean; q += dv * dv; }
#pragma unroll
  for (int d = 16; d >= 1; d >>= 1) q += __shfl_xor(q, d, 32);
  float rstd = rsqrtf(q * (1.0f / 384.0f) + 1e-5f);
  half_t* dst = out + t * 384;
#pragma unroll
  for (int j = 0; j < 12; j++) {
    int c = lane + 32 * j;
    dst[c] = (half_t)((v[j] - mean) * rstd * gam[c] + bet[c]);
  }
}

// ---------------------------------------------------------------------------
// WMMA GEMM: C[M,N] = A[M,K](f16) * Bt[N,K]^T (f16) + bias.  Epilogues:
// EPI 0: f16 out       EPI 1: f16 + exact GELU
// EPI 2: f32, un-window + roll(+3) + residual (proj)
// EPI 3: f32, natural order + residual (fc2)
// Block: 128 thr = 4 waves (2x2); wave tile 64x64; k-step 32; double-buffered
// A tile staged by TDM (TENSORcnt) or async-to-LDS (ASYNCcnt).
// ---------------------------------------------------------------------------
template <int EPI>
__global__ __launch_bounds__(128) void gemm_wmma(const half_t* __restrict__ A,
                                                 const half_t* __restrict__ Bt,
                                                 const float* __restrict__ bias,
                                                 half_t* __restrict__ outh,
                                                 float* outf, const float* res,
                                                 int M, int N, int K) {
  __shared__ half_t As[2][128 * 40];   // 128 rows x 32 halves, stride 40
  const int tid = threadIdx.x;
  const int lane = tid & 31, wave = tid >> 5;
  const int wm = wave >> 1, wn = wave & 1;
  const size_t bm = (size_t)blockIdx.x * 128;
  const int bn = blockIdx.y * 128;

  v8f acc[4][4];
#pragma unroll
  for (int i = 0; i < 4; i++)
#pragma unroll
    for (int j = 0; j < 4; j++) acc[i][j] = (v8f){};

  const half_t* Atile = A + bm * (size_t)K;       // block's 128-row slab
  const half_t* Arow = Atile + (size_t)tid * K;   // this thread's row

#if STAGE_TDM
  if (tid == 0) tdm_load_tile(Atile, &As[0][0], K);
#elif STAGE_ASYNC
  {
    half_t* my = &As[0][tid * 40];
#pragma unroll
    for (int c = 0; c < 4; c++) async_copy16(Arow + c * 8, my + c * 8);
  }
#endif

  int cur = 0;
  for (int k0 = 0; k0 < K; k0 += 32) {
#if STAGE_TDM
    __syncthreads();                       // everyone done reading buf being refilled
    if (tid == 0) {
      if (k0 + 32 < K) {
        tdm_load_tile(Atile + k0 + 32, &As[cur ^ 1][0], K);
        __builtin_amdgcn_s_wait_tensorcnt(1);   // current tile done, next in flight
      } else {
        __builtin_amdgcn_s_wait_tensorcnt(0);
      }
    }
    __syncthreads();                       // current buffer published to all waves
#elif STAGE_ASYNC
    WAIT_ASYNC0();                         // my async writes for As[cur] done
    __syncthreads();                       // everyone's writes done; readers of
                                           // As[cur^1] from prev iter also done
    if (k0 + 32 < K) {
      half_t* my = &As[cur ^ 1][tid * 40];
#pragma unroll
      for (int c = 0; c < 4; c++) async_copy16(Arow + k0 + 32 + c * 8, my + c * 8);
    }
#else
    __builtin_prefetch(Arow + k0 + 128, 0, 1);
    v8h t0 = *(const v8h*)(Arow + k0);
    v8h t1 = *(const v8h*)(Arow + k0 + 8);
    v8h t2 = *(const v8h*)(Arow + k0 + 16);
    v8h t3 = *(const v8h*)(Arow + k0 + 24);
    __syncthreads();
    v8h* dst = (v8h*)(&As[0][(size_t)tid * 40]);
    dst[0] = t0; dst[1] = t1; dst[2] = t2; dst[3] = t3;
    __syncthreads();
#endif

    v16h bf[4], af[4];
#pragma unroll
    for (int j = 0; j < 4; j++)
      bf[j] = load_b_frag(Bt + (size_t)(bn + wn * 64 + j * 16) * K + k0, K, lane);
#pragma unroll
    for (int i = 0; i < 4; i++)
      af[i] = load_a_frag(&As[cur][(wm * 64 + i * 16) * 40], 40, lane);
#pragma unroll
    for (int i = 0; i < 4; i++)
#pragma unroll
      for (int j = 0; j < 4; j++)
        acc[i][j] = WMMA_F16(af[i], bf[j], acc[i][j]);

#if STAGE_TDM || STAGE_ASYNC
    cur ^= 1;
#endif
  }

  const int hf = lane >> 4, ln = lane & 15;
#pragma unroll
  for (int i = 0; i < 4; i++) {
#pragma unroll
    for (int j = 0; j < 4; j++) {
      int coln = bn + wn * 64 + j * 16 + ln;
      float bj = bias[coln];
#pragma unroll
      for (int r = 0; r < 8; r++) {
        size_t row = bm + wm * 64 + i * 16 + hf * 8 + r;
        float v = acc[i][j][r] + bj;
        if (EPI == 0) {
          outh[row * (size_t)N + coln] = (half_t)v;
        } else if (EPI == 1) {
          float g = 0.5f * v * (1.0f + erff(v * 0.70710678118654752f));
          outh[row * (size_t)N + coln] = (half_t)g;
        } else if (EPI == 2) {
          int wt = (int)row;
          int win = wt / 49, ii = wt - win * 49;
          int b = win >> 6, w = win & 63;
          int r7 = ii / 7, c7 = ii - r7 * 7;
          int gh = (w >> 3) * 7 + r7, gw = (w & 7) * 7 + c7;
          int oh = gh + 3; if (oh >= 56) oh -= 56;   // inverse roll(+3)
          int ow = gw + 3; if (ow >= 56) ow -= 56;
          size_t idx = ((size_t)b * 3136 + oh * 56 + ow) * 384 + coln;
          outf[idx] = res[idx] + v;
        } else {
          size_t idx = row * (size_t)N + coln;
          outf[idx] = res[idx] + v;
        }
      }
    }
  }
}

// ---------------------------------------------------------------------------
// Window attention: one wave per (window, head). N=49 padded to 64, hd=32.
// ---------------------------------------------------------------------------
__global__ __launch_bounds__(128) void attn_kernel(const half_t* __restrict__ qkv,
                                                   const float* __restrict__ btab,
                                                   half_t* __restrict__ outa) {
  __shared__ half_t sm[4 * (64 * 72 + 32 * 80)];  // per-wave: S(64x72) + Vt(32x80)
  const int wave = threadIdx.x >> 5, lane = threadIdx.x & 31;
  half_t* Ss = sm + wave * (64 * 72 + 32 * 80);
  half_t* Vt = Ss + 64 * 72;

  const int pair = blockIdx.x * 4 + wave;         // 49152 pairs
  const int win = pair / 12, h = pair - win * 12;
  const half_t* Qb = qkv + (size_t)win * 49 * 1152 + h * 32;  // +0:q +384:k +768:v

  // ---- stage V transposed into LDS (dims x tokens), zero padded cols ----
  v8h z = {};
  for (int p = lane; p < (32 * 80) / 8; p += 32) ((v8h*)Vt)[p] = z;
  for (int t = lane; t < 49; t += 32) {
    const half_t* vp = Qb + 768 + (size_t)t * 1152;
#pragma unroll
    for (int d = 0; d < 32; d++) Vt[d * 80 + t] = vp[d];
  }

  // ---- S = Q K^T via WMMA ----
  v8f acc[4][4];
#pragma unroll
  for (int i = 0; i < 4; i++)
#pragma unroll
    for (int j = 0; j < 4; j++) acc[i][j] = (v8f){};

  const size_t maxrow = (size_t)MTOK - 1;
  v16h af[4], bf[4];
#pragma unroll
  for (int i = 0; i < 4; i++) {
    size_t row = (size_t)win * 49 + i * 16 + (lane & 15);
    if (row > maxrow) row = maxrow;
    af[i] = load_a_frag_ptr(qkv + row * 1152 + h * 32, lane);
  }
#pragma unroll
  for (int j = 0; j < 4; j++) {
    size_t row = (size_t)win * 49 + j * 16 + (lane & 15);
    if (row > maxrow) row = maxrow;
    bf[j] = load_b_frag_ptr(qkv + row * 1152 + 384 + h * 32, lane);
  }
#pragma unroll
  for (int i = 0; i < 4; i++)
#pragma unroll
    for (int j = 0; j < 4; j++)
      acc[i][j] = WMMA_F16(af[i], bf[j], acc[i][j]);

  // ---- scale + rel-pos bias + shifted-window mask ----
  const float scale = 0.17677669529663689f;   // 1/sqrt(32)
  const int hf = lane >> 4, ln = lane & 15;
  const int wimg = win & 63, wh = wimg >> 3, ww = wimg & 7;

  int regj[4]; bool valj[4];
#pragma unroll
  for (int j = 0; j < 4; j++) {
    int tj = j * 16 + ln;
    valj[j] = tj < 49;
    int r7 = tj / 7, c7 = tj - r7 * 7;
    int gh = wh * 7 + r7, gw = ww * 7 + c7;
    int hg = gh < 49 ? 0 : (gh < 53 ? 1 : 2);
    int wg = gw < 49 ? 0 : (gw < 53 ? 1 : 2);
    regj[j] = hg * 3 + wg;
  }
#pragma unroll
  for (int i = 0; i < 4; i++) {
#pragma unroll
    for (int r = 0; r < 8; r++) {
      int ti = i * 16 + hf * 8 + r;
      bool vali = ti < 49;
      int r7 = ti / 7, c7 = ti - r7 * 7;
      int gh = wh * 7 + r7, gw = ww * 7 + c7;
      int hg = gh < 49 ? 0 : (gh < 53 ? 1 : 2);
      int wg = gw < 49 ? 0 : (gw < 53 ? 1 : 2);
      int regi = hg * 3 + wg;
      const float* tr = btab + ((size_t)h * 49 + (vali ? ti : 0)) * 49;
#pragma unroll
      for (int j = 0; j < 4; j++) {
        float s;
        if (vali && valj[j])
          s = acc[i][j][r] * scale + tr[j * 16 + ln] + ((regi != regj[j]) ? -100.f : 0.f);
        else
          s = -1e30f;
        acc[i][j][r] = s;
      }
    }
  }

  // ---- row softmax ----
#pragma unroll
  for (int i = 0; i < 4; i++) {
#pragma unroll
    for (int r = 0; r < 8; r++) {
      float m = acc[i][0][r];
#pragma unroll
      for (int j = 1; j < 4; j++) m = fmaxf(m, acc[i][j][r]);
#pragma unroll
      for (int d = 1; d < 16; d <<= 1) m = fmaxf(m, __shfl_xor(m, d, 32));
      float s = 0.f;
#pragma unroll
      for (int j = 0; j < 4; j++) {
        float e = __expf(acc[i][j][r] - m);
        acc[i][j][r] = e;
        s += e;
      }
#pragma unroll
      for (int d = 1; d < 16; d <<= 1) s += __shfl_xor(s, d, 32);
      float rinv = 1.0f / s;
#pragma unroll
      for (int j = 0; j < 4; j++) acc[i][j][r] *= rinv;
    }
  }

  // ---- P -> LDS (re-fragment as A), then O = P V ----
#pragma unroll
  for (int i = 0; i < 4; i++)
#pragma unroll
    for (int j = 0; j < 4; j++)
#pragma unroll
      for (int r = 0; r < 8; r++)
        Ss[(i * 16 + hf * 8 + r) * 72 + j * 16 + ln] = (half_t)acc[i][j][r];

  v8f oacc[4][2];
#pragma unroll
  for (int i = 0; i < 4; i++) { oacc[i][0] = (v8f){}; oacc[i][1] = (v8f){}; }
#pragma unroll
  for (int ks = 0; ks < 64; ks += 32) {
    v16h pa[4], vb[2];
#pragma unroll
    for (int i = 0; i < 4; i++) pa[i] = load_a_frag(Ss + (i * 16) * 72 + ks, 72, lane);
#pragma unroll
    for (int n = 0; n < 2; n++) vb[n] = load_b_frag(Vt + (n * 16) * 80 + ks, 80, lane);
#pragma unroll
    for (int i = 0; i < 4; i++)
#pragma unroll
      for (int n = 0; n < 2; n++)
        oacc[i][n] = WMMA_F16(pa[i], vb[n], oacc[i][n]);
  }

  half_t* ob = outa + (size_t)win * 49 * 384 + h * 32;
#pragma unroll
  for (int i = 0; i < 4; i++)
#pragma unroll
    for (int r = 0; r < 8; r++) {
      int ti = i * 16 + hf * 8 + r;
      if (ti < 49) {
#pragma unroll
        for (int n = 0; n < 2; n++)
          ob[(size_t)ti * 384 + n * 16 + ln] = (half_t)oacc[i][n][r];
      }
    }
}

// ---------------------------------------------------------------------------
extern "C" void kernel_launch(void* const* d_in, const int* in_sizes, int n_in,
                              void* d_out, int out_size, void* d_ws, size_t ws_size,
                              hipStream_t stream) {
  const float* x      = (const float*)d_in[0];
  const float* n1g    = (const float*)d_in[1];
  const float* n1b    = (const float*)d_in[2];
  const float* qkv_w  = (const float*)d_in[3];
  const float* qkv_b  = (const float*)d_in[4];
  const float* rpb    = (const float*)d_in[5];
  const float* proj_w = (const float*)d_in[6];
  const float* proj_b = (const float*)d_in[7];
  const float* n2g    = (const float*)d_in[8];
  const float* n2b    = (const float*)d_in[9];
  const float* fc1_w  = (const float*)d_in[10];
  const float* fc1_b  = (const float*)d_in[11];
  const float* fc2_w  = (const float*)d_in[12];
  const float* fc2_b  = (const float*)d_in[13];
  float* out = (float*)d_out;

  char* ws = (char*)d_ws;
  size_t off = 0;
  auto take = [&](size_t bytes) {
    char* p = ws + off;
    off = (off + bytes + 255) & ~(size_t)255;
    return p;
  };
  half_t* wqkvT  = (half_t*)take((size_t)1152 * 384 * 2);
  half_t* wprojT = (half_t*)take((size_t)384 * 384 * 2);
  half_t* wfc1T  = (half_t*)take((size_t)1536 * 384 * 2);
  half_t* wfc2T  = (half_t*)take((size_t)384 * 1536 * 2);
  float*  btab   = (float*)take((size_t)12 * 49 * 49 * 4);
  half_t* R3 = (half_t*)take((size_t)MTOK * 384 * 2);    // xw / attn_out / h2
  half_t* R1 = (half_t*)take((size_t)MTOK * 1536 * 2);   // qkv (1152 cols) / fc1 act

  // weight prep
  convT_kernel<<<(384 * 1152 + 255) / 256, 256, 0, stream>>>(qkv_w, wqkvT, 384, 1152);
  convT_kernel<<<(384 * 384 + 255) / 256, 256, 0, stream>>>(proj_w, wprojT, 384, 384);
  convT_kernel<<<(384 * 1536 + 255) / 256, 256, 0, stream>>>(fc1_w, wfc1T, 384, 1536);
  convT_kernel<<<(1536 * 384 + 255) / 256, 256, 0, stream>>>(fc2_w, wfc2T, 1536, 384);
  biastab_kernel<<<(12 * 49 * 49 + 255) / 256, 256, 0, stream>>>(rpb, btab);

  // LN1 + roll + window partition -> f16
  ln_kernel<<<MTOK / 8, 256, 0, stream>>>(x, n1g, n1b, R3, 0);

  // qkv GEMM: (200704 x 384) x (384 x 1152)
  gemm_wmma<0><<<dim3(MTOK / 128, 1152 / 128), 128, 0, stream>>>(
      R3, wqkvT, qkv_b, R1, nullptr, nullptr, MTOK, 1152, 384);

  // window attention (4096 windows * 12 heads, 4 wave-tasks per block)
  attn_kernel<<<(4096 * 12) / 4, 128, 0, stream>>>(R1, btab, R3);

  // proj GEMM + un-window + inverse roll + residual -> x_new (in d_out, f32)
  gemm_wmma<2><<<dim3(MTOK / 128, 384 / 128), 128, 0, stream>>>(
      R3, wprojT, proj_b, nullptr, out, x, MTOK, 384, 384);

  // LN2 -> f16
  ln_kernel<<<MTOK / 8, 256, 0, stream>>>(out, n2g, n2b, R3, 1);

  // fc1 GEMM + GELU
  gemm_wmma<1><<<dim3(MTOK / 128, 1536 / 128), 128, 0, stream>>>(
      R3, wfc1T, fc1_b, R1, nullptr, nullptr, MTOK, 1536, 384);

  // fc2 GEMM + residual (x_new read from d_out, final written to d_out)
  gemm_wmma<3><<<dim3(MTOK / 128, 384 / 128), 128, 0, stream>>>(
      R1, wfc2T, fc2_b, nullptr, out, (const float*)out, MTOK, 384, 1536);
}
